// LongformerClassifier_70806830842652
// MI455X (gfx1250) — compile-verified, matrix operand order checked by reading, not run on hardware
//
#include <hip/hip_runtime.h>
#include <hip/hip_bf16.h>
#include <math.h>
#include <stdint.h>

// ---- model dims ----
#define LAYERS  12
#define HIDDEN  768
#define NHEADS  12
#define HEADD   64
#define FFDIM   3072
#define BATCH   2
#define SEQ     2048
#define WWIN    256
#define NTOK    (BATCH*SEQ)
#define LNEPS   1e-5f

typedef _Float16 v16h __attribute__((ext_vector_type(16)));
typedef float    v8f  __attribute__((ext_vector_type(8)));
typedef unsigned int u32x4 __attribute__((ext_vector_type(4)));
typedef int          i32x4 __attribute__((ext_vector_type(4)));
typedef int          i32x8 __attribute__((ext_vector_type(8)));

__device__ __forceinline__ v8f wmma16x16x32(v16h a, v16h b, v8f c) {
  // D = A(16x32 f16) x B(32x16 f16) + C(16x16 f32)
  return __builtin_amdgcn_wmma_f32_16x16x32_f16(false, a, false, b, (short)0, c,
                                                false, false);
}

// K index within a 32-wide fragment for dword-group g (pairs are contiguous):
// groups 0..3 -> K = {0,2,4,6}, groups 4..7 -> K = {16,18,20,22}; +8 for hi half.
__device__ __forceinline__ constexpr int kmap(int g) {
  return (g < 4) ? (g << 1) : (16 + ((g - 4) << 1));
}

// Fragment where the K-dimension runs along a row (pairs contiguous in memory).
// Used for A (m = lane%16) and for B when the tile is stored [n][k].
__device__ __forceinline__ v16h load_frag_rowK(const _Float16* base, int row0,
                                               int pitch, int col0, int lane) {
  union { v16h v; unsigned u[8]; } f;
  int hi = (lane >> 4) & 1;
  int row = row0 + (lane & 15);
  const _Float16* p = base + (size_t)row * pitch + col0;
#pragma unroll
  for (int g = 0; g < 8; ++g)
    f.u[g] = *(const unsigned*)(p + kmap(g) + 8 * hi);
  return f.v;
}

// B fragment from a tile stored [k][n] (K runs across rows): scalar gathers.
__device__ __forceinline__ v16h load_frag_colN(const _Float16* base, int n0,
                                               int pitch, int lane) {
  v16h f;
  int hi = (lane >> 4) & 1;
  int n = n0 + (lane & 15);
#pragma unroll
  for (int e = 0; e < 16; ++e) {
    int kk = kmap(e >> 1) + 8 * hi + (e & 1);
    f[e] = base[(size_t)kk * pitch + n];
  }
  return f;
}

// ---------------------------------------------------------------------------
// Tensor Data Mover: async 2D f16 tile, global -> LDS, optional LDS row pad.
// D# packing per CDNA5 ISA ch.8 (group0 128b, group1 256b, groups2/3 = 0).
//   width/rows  : tile_dim0 / tile_dim1 (and tensor dims == tile -> no OOB)
//   stride_elems: tensor_dim0_stride (elements)
//   pad codes   : interval 0=2dw,1=4,2=8,3=16,4=32...; amount code = dwords-1
//                 (pad_amt_code < 0 -> no padding)
// This toolchain's builtin is the 6-arg form:
//   (u32x4 g0, i32x8 g1, i32x4 g2, i32x4 g3, i32x8 extra, i32 cpol)
// ---------------------------------------------------------------------------
__device__ __forceinline__ void tdm_load_2d_f16(unsigned lds_byte,
                                                const _Float16* gaddr,
                                                unsigned width, unsigned rows,
                                                unsigned long long stride_elems,
                                                int pad_int_code,
                                                int pad_amt_code) {
  unsigned long long ga = (unsigned long long)(uintptr_t)gaddr;
  u32x4 g0;
  g0[0] = 1u;                                     // count=1 (valid), user mode
  g0[1] = lds_byte;                               // lds_addr (bytes)
  g0[2] = (unsigned)(ga & 0xffffffffu);           // global_addr[31:0]
  g0[3] = (unsigned)((ga >> 32) & 0x01ffffffu)    // global_addr[56:32]
        | (2u << 30);                             // type = 2 ("image")
  unsigned d0 = (1u << 16);                       // data_size = 1 -> 2 bytes
  if (pad_amt_code >= 0)
    d0 |= (1u << 20) | ((unsigned)pad_int_code << 22)
        | ((unsigned)pad_amt_code << 25);
  i32x8 g1;
  g1[0] = (int)d0;                                // mask=0, flags
  g1[1] = (int)((width & 0xffffu) << 16);         // tensor_dim0[15:0] (bits63:48)
  g1[2] = (int)((width >> 16) | ((rows & 0xffffu) << 16));   // dim0 hi | dim1 lo
  g1[3] = (int)((rows >> 16) | ((width & 0xffffu) << 16));   // dim1 hi | tile_dim0
  g1[4] = (int)(rows & 0xffffu);                  // tile_dim1 (tile_dim2 = 0)
  g1[5] = (int)(unsigned)(stride_elems & 0xffffffffu);       // dim0_stride lo
  g1[6] = (int)(unsigned)((stride_elems >> 32) & 0xffffu);   // dim0_stride hi
  g1[7] = 0;                                      // tensor_dim1_stride hi = 0
  i32x4 z4 = {0, 0, 0, 0};
  i32x8 z8 = {0, 0, 0, 0, 0, 0, 0, 0};
  __builtin_amdgcn_tensor_load_to_lds(g0, g1, z4, z4, z8, 0);
}

// ---------------------------------------------------------------------------
// position ids: cumsum(mask)*mask + 1
// ---------------------------------------------------------------------------
__global__ void pos_kernel(const int* __restrict__ amask, int* __restrict__ pos_ids) {
  int b = threadIdx.x;
  if (b >= BATCH) return;
  int run = 0;
  for (int s = 0; s < SEQ; ++s) {
    int m = amask[b * SEQ + s];
    run += m;
    pos_ids[b * SEQ + s] = run * m + 1;
  }
}

// ---------------------------------------------------------------------------
// fp32 -> f16 weight conversion (8 elems/thread, exact grids)
// ---------------------------------------------------------------------------
__global__ __launch_bounds__(256)
void cvt_kernel(const float* __restrict__ src, _Float16* __restrict__ dst) {
  size_t i = ((size_t)blockIdx.x * 256 + threadIdx.x) * 8;
  float4 a = *(const float4*)(src + i);
  float4 b = *(const float4*)(src + i + 4);
  union { _Float16 h[8]; uint4 u; } p;
  p.h[0] = (_Float16)a.x; p.h[1] = (_Float16)a.y;
  p.h[2] = (_Float16)a.z; p.h[3] = (_Float16)a.w;
  p.h[4] = (_Float16)b.x; p.h[5] = (_Float16)b.y;
  p.h[6] = (_Float16)b.z; p.h[7] = (_Float16)b.w;
  *(uint4*)(dst + i) = p.u;
}

// ---------------------------------------------------------------------------
// embeddings + LayerNorm -> h (f32) and h (f16)
// ---------------------------------------------------------------------------
__global__ __launch_bounds__(256)
void embed_kernel(const int* __restrict__ ids, const int* __restrict__ pos_ids,
                  const float* __restrict__ wemb, const float* __restrict__ pemb,
                  const float* __restrict__ temb, const float* __restrict__ lw,
                  const float* __restrict__ lb, float* __restrict__ h32,
                  _Float16* __restrict__ h16) {
  __shared__ float red[256];
  int tok = blockIdx.x, t = threadIdx.x;
  int id = ids[tok], pid = pos_ids[tok];
  float x[3];
#pragma unroll
  for (int i = 0; i < 3; ++i) {
    int c = t + i * 256;
    x[i] = wemb[(size_t)id * HIDDEN + c] + pemb[(size_t)pid * HIDDEN + c] + temb[c];
  }
  red[t] = x[0] + x[1] + x[2];
  __syncthreads();
  for (int o = 128; o > 0; o >>= 1) { if (t < o) red[t] += red[t + o]; __syncthreads(); }
  float mean = red[0] / (float)HIDDEN;
  __syncthreads();
  float vs = 0.f;
#pragma unroll
  for (int i = 0; i < 3; ++i) { float d = x[i] - mean; vs += d * d; }
  red[t] = vs;
  __syncthreads();
  for (int o = 128; o > 0; o >>= 1) { if (t < o) red[t] += red[t + o]; __syncthreads(); }
  float rstd = rsqrtf(red[0] / (float)HIDDEN + LNEPS);
#pragma unroll
  for (int i = 0; i < 3; ++i) {
    int c = t + i * 256;
    float y = (x[i] - mean) * rstd * lw[c] + lb[c];
    h32[(size_t)tok * HIDDEN + c] = y;
    h16[(size_t)tok * HIDDEN + c] = (_Float16)y;
  }
}

// ---------------------------------------------------------------------------
// residual add + LayerNorm -> h (f32) and h (f16)   (h32 may alias hin)
// ---------------------------------------------------------------------------
__global__ __launch_bounds__(256)
void ln_add_kernel(const float* hin, const float* addin,
                   const float* __restrict__ lw, const float* __restrict__ lb,
                   float* h32, _Float16* __restrict__ h16) {
  __shared__ float red[256];
  int tok = blockIdx.x, t = threadIdx.x;
  float x[3];
#pragma unroll
  for (int i = 0; i < 3; ++i) {
    int c = t + i * 256;
    x[i] = hin[(size_t)tok * HIDDEN + c] + addin[(size_t)tok * HIDDEN + c];
  }
  red[t] = x[0] + x[1] + x[2];
  __syncthreads();
  for (int o = 128; o > 0; o >>= 1) { if (t < o) red[t] += red[t + o]; __syncthreads(); }
  float mean = red[0] / (float)HIDDEN;
  __syncthreads();
  float vs = 0.f;
#pragma unroll
  for (int i = 0; i < 3; ++i) { float d = x[i] - mean; vs += d * d; }
  red[t] = vs;
  __syncthreads();
  for (int o = 128; o > 0; o >>= 1) { if (t < o) red[t] += red[t + o]; __syncthreads(); }
  float rstd = rsqrtf(red[0] / (float)HIDDEN + LNEPS);
#pragma unroll
  for (int i = 0; i < 3; ++i) {
    int c = t + i * 256;
    float y = (x[i] - mean) * rstd * lw[c] + lb[c];
    h32[(size_t)tok * HIDDEN + c] = y;
    h16[(size_t)tok * HIDDEN + c] = (_Float16)y;
  }
}

// ---------------------------------------------------------------------------
// GEMM: out[m,n] = act( A[m,:] . W[n,:] + bias[n] ),  A f16 [M,K], W f16 [N,K].
//   Block tile 128x64, BK=32, 8 waves, each wave a 32x32 macro tile (2x2 WMMA).
//   TDM double-buffered LDS staging: wave 0 issues tensor_load_to_lds for tile
//   k+1 while all waves compute on tile k (TENSORcnt pipelining).
//   mode 0: f32 out [M,N]; mode 2: exact GELU -> f16 out [M,N];
//   mode 3: f16 out scaled, written transposed to [B,NH,S,HD] (QKV).
// ---------------------------------------------------------------------------
#define BM 128
#define BN 64
#define BK 32
#define GPIT 40   // LDS row pitch in halves: 16dw row + 4dw pad via TDM padding

__global__ __launch_bounds__(256)
void gemm_kernel(const _Float16* __restrict__ A, const _Float16* __restrict__ Wt,
                 const float* __restrict__ bias, int M, int N, int K,
                 int mode, float scale, float* __restrict__ out32,
                 _Float16* __restrict__ out16) {
  __shared__ __align__(16) _Float16 sA[2 * BM * GPIT];
  __shared__ __align__(16) _Float16 sB[2 * BN * GPIT];
  int t = threadIdx.x, lane = t & 31, wave = t >> 5;
  int hi = (lane >> 4) & 1;
  int wm = wave & 3, wn = wave >> 2;               // 4 x 2 wave grid
  int gm0 = blockIdx.y * BM, gn0 = blockIdx.x * BN;

  unsigned ldsA[2], ldsB[2];
  ldsA[0] = (unsigned)(uintptr_t)(void*)&sA[0];
  ldsA[1] = (unsigned)(uintptr_t)(void*)&sA[BM * GPIT];
  ldsB[0] = (unsigned)(uintptr_t)(void*)&sB[0];
  ldsB[1] = (unsigned)(uintptr_t)(void*)&sB[BN * GPIT];

  v8f zero8;
#pragma unroll
  for (int r = 0; r < 8; ++r) zero8[r] = 0.f;
  v8f c[2][2];
  c[0][0] = zero8; c[0][1] = zero8; c[1][0] = zero8; c[1][1] = zero8;

  int ksteps = K / BK;

  // prologue: stage tile 0 (rows x 32 halves, 16dw + 4dw pad -> pitch 40)
  if (wave == 0) {
    tdm_load_2d_f16(ldsA[0], A + (size_t)gm0 * K, BK, BM, (unsigned long long)K, 3, 3);
    tdm_load_2d_f16(ldsB[0], Wt + (size_t)gn0 * K, BK, BN, (unsigned long long)K, 3, 3);
  }

  for (int kb = 0; kb < ksteps; ++kb) {
    int buf = kb & 1;
    if (kb + 1 < ksteps) {
      if (wave == 0) {
        tdm_load_2d_f16(ldsA[buf ^ 1], A + (size_t)gm0 * K + (kb + 1) * BK,
                        BK, BM, (unsigned long long)K, 3, 3);
        tdm_load_2d_f16(ldsB[buf ^ 1], Wt + (size_t)gn0 * K + (kb + 1) * BK,
                        BK, BN, (unsigned long long)K, 3, 3);
        __builtin_amdgcn_s_wait_tensorcnt((short)2);   // tile kb landed
      }
    } else if (wave == 0) {
      __builtin_amdgcn_s_wait_tensorcnt((short)0);
    }
    __syncthreads();

    const _Float16* curA = sA + buf * (BM * GPIT);
    const _Float16* curB = sB + buf * (BN * GPIT);
    v16h a0 = load_frag_rowK(curA, wm * 32,      GPIT, 0, lane);
    v16h a1 = load_frag_rowK(curA, wm * 32 + 16, GPIT, 0, lane);
    v16h b0 = load_frag_rowK(curB, wn * 32,      GPIT, 0, lane);
    v16h b1 = load_frag_rowK(curB, wn * 32 + 16, GPIT, 0, lane);
    c[0][0] = wmma16x16x32(a0, b0, c[0][0]);
    c[0][1] = wmma16x16x32(a0, b1, c[0][1]);
    c[1][0] = wmma16x16x32(a1, b0, c[1][0]);
    c[1][1] = wmma16x16x32(a1, b1, c[1][1]);
    __syncthreads();   // all reads of buf done before TDM reuses it
  }

#pragma unroll
  for (int i = 0; i < 2; ++i) {
#pragma unroll
    for (int j = 0; j < 2; ++j) {
      int gn = gn0 + wn * 32 + j * 16 + (lane & 15);
      float bsv = bias[gn];
#pragma unroll
      for (int r = 0; r < 8; ++r) {
        int gm = gm0 + wm * 32 + i * 16 + r + 8 * hi;
        float vv = c[i][j][r] + bsv;
        if (mode == 0) {
          out32[(size_t)gm * N + gn] = vv;
        } else if (mode == 2) {
          float g = 0.5f * vv * (1.f + erff(vv * 0.70710678118654752f));
          out16[(size_t)gm * N + gn] = (_Float16)g;
        } else {                                    // mode 3: QKV transpose write
          int bb = gm / SEQ, ss = gm - bb * SEQ;
          int head = gn / HEADD, hd = gn - head * HEADD;
          out16[(((size_t)(bb * NHEADS + head)) * SEQ + ss) * HEADD + hd] =
              (_Float16)(vv * scale);
        }
      }
    }
  }
}

// ---------------------------------------------------------------------------
// Sliding-window attention, flash-style online softmax.
//   Block = 128 threads = 4 waves; each wave owns 32 queries and its exact key
//   window [q0-W, q0+32+W) in steps of 32.  Per-wave LDS K/V/P tiles; no block
//   barriers (trip counts differ per wave) -> explicit s_wait_dscnt ordering.
// ---------------------------------------------------------------------------
#define APITCH 72   // K/V tile pitch in halves (144B rows, 16B aligned)
#define PPITCH 40

__global__ __launch_bounds__(128)
void attn_kernel(const _Float16* __restrict__ qm, const _Float16* __restrict__ km,
                 const _Float16* __restrict__ vm, const int* __restrict__ amask,
                 _Float16* __restrict__ attn_out) {
  __shared__ __align__(16) _Float16 sK[4][32 * APITCH];
  __shared__ __align__(16) _Float16 sV[4][32 * APITCH];
  __shared__ __align__(16) _Float16 sP[4][32 * PPITCH];

  int t = threadIdx.x, lane = t & 31, wave = t >> 5;
  int hi = (lane >> 4) & 1;
  int b = blockIdx.z, h = blockIdx.y;
  int q0 = blockIdx.x * 128 + wave * 32;           // multiple of 32
  size_t headbase = ((size_t)(b * NHEADS + h)) * SEQ * HEADD;

  _Float16* Kt = sK[wave];
  _Float16* Vt = sV[wave];
  _Float16* Pt = sP[wave];
  const int* mrow = amask + b * SEQ;

  // Q fragments: 2 m-subtiles x 2 k-halves of HEADD (q already scaled by 1/8)
  v16h qf[2][2];
#pragma unroll
  for (int mi = 0; mi < 2; ++mi)
#pragma unroll
    for (int hh = 0; hh < 2; ++hh)
      qf[mi][hh] = load_frag_rowK(qm + headbase + (size_t)(q0 + mi * 16) * HEADD,
                                  0, HEADD, hh * 32, lane);

  v8f zero8;
#pragma unroll
  for (int r = 0; r < 8; ++r) zero8[r] = 0.f;
  v8f o[2][4];
#pragma unroll
  for (int mi = 0; mi < 2; ++mi)
#pragma unroll
    for (int tt = 0; tt < 4; ++tt) o[mi][tt] = zero8;

  float mrun[2][8], lrun[2][8];
#pragma unroll
  for (int mi = 0; mi < 2; ++mi)
#pragma unroll
    for (int r = 0; r < 8; ++r) { mrun[mi][r] = -3e38f; lrun[mi][r] = 0.f; }

  int klo = q0 - WWIN; if (klo < 0) klo = 0;       // stays 32-aligned
  int khi = q0 + 32 + WWIN; if (khi > SEQ) khi = SEQ;

  for (int ks = klo; ks < khi; ks += 32) {
    // stage K/V tiles: 4x b128 (first 32 of 64 halves) of one key row per lane
    {
      int key = ks + lane;
      const uint4* kg = (const uint4*)(km + headbase + (size_t)key * HEADD);
      const uint4* vg = (const uint4*)(vm + headbase + (size_t)key * HEADD);
      uint4* kd = (uint4*)(Kt + lane * APITCH);
      uint4* vd = (uint4*)(Vt + lane * APITCH);
#pragma unroll
      for (int i = 0; i < 4; ++i) { kd[i] = kg[i]; vd[i] = vg[i]; }
      if (ks + 32 < khi) {                         // stream next tiles into GL2
        __builtin_prefetch(km + headbase + (size_t)(key + 32) * HEADD, 0, 1);
        __builtin_prefetch(vm + headbase + (size_t)(key + 32) * HEADD, 0, 1);
      }
    }
    asm volatile("s_wait_dscnt 0" ::: "memory");   // wave-local LDS ordering

    int n0 = ks + (lane & 15), n1 = n0 + 16;
    int mv0 = mrow[n0], mv1 = mrow[n1];

    // B fragments for S = Q . K^T  (Kt is [key][hd] -> rowK layout)
    v16h bk[2][2];
#pragma unroll
    for (int j = 0; j < 2; ++j)
#pragma unroll
      for (int hh = 0; hh < 2; ++hh)
        bk[j][hh] = load_frag_rowK(Kt, j * 16, APITCH, hh * 32, lane);

#pragma unroll
    for (int mi = 0; mi < 2; ++mi) {
      v8f c0 = zero8, c1 = zero8;
      c0 = wmma16x16x32(qf[mi][0], bk[0][0], c0);
      c0 = wmma16x16x32(qf[mi][1], bk[0][1], c0);
      c1 = wmma16x16x32(qf[mi][0], bk[1][0], c1);
      c1 = wmma16x16x32(qf[mi][1], bk[1][1], c1);

#pragma unroll
      for (int r = 0; r < 8; ++r) {
        int row = q0 + mi * 16 + r + 8 * hi;
        int d0 = n0 - row; if (d0 < 0) d0 = -d0;
        int d1 = n1 - row; if (d1 < 0) d1 = -d1;
        float s0 = (d0 <= WWIN && mv0 != 0) ? c0[r] : -1e9f;
        float s1 = (d1 <= WWIN && mv1 != 0) ? c1[r] : -1e9f;
        float rm = fmaxf(s0, s1);
#pragma unroll
        for (int off = 1; off < 16; off <<= 1)
          rm = fmaxf(rm, __shfl_xor(rm, off, 32));
        float mn = fmaxf(mrun[mi][r], rm);
        float al = expf(mrun[mi][r] - mn);
        float p0 = expf(s0 - mn), p1 = expf(s1 - mn);
        float rs = p0 + p1;
#pragma unroll
        for (int off = 1; off < 16; off <<= 1)
          rs += __shfl_xor(rs, off, 32);
        mrun[mi][r] = mn;
        lrun[mi][r] = lrun[mi][r] * al + rs;
        int prow = mi * 16 + r + 8 * hi;
        Pt[prow * PPITCH + (lane & 15)]      = (_Float16)p0;
        Pt[prow * PPITCH + 16 + (lane & 15)] = (_Float16)p1;
#pragma unroll
        for (int tt = 0; tt < 4; ++tt) o[mi][tt][r] *= al;
      }
    }
    asm volatile("s_wait_dscnt 0" ::: "memory");

    // O += P . V    (P re-fragmented as A 16x32; Vt is [key][hd] -> colN layout)
    v16h pa0 = load_frag_rowK(Pt, 0,  PPITCH, 0, lane);
    v16h pa1 = load_frag_rowK(Pt, 16, PPITCH, 0, lane);
#pragma unroll
    for (int tt = 0; tt < 4; ++tt) {
      v16h bv = load_frag_colN(Vt, tt * 16, APITCH, lane);
      o[0][tt] = wmma16x16x32(pa0, bv, o[0][tt]);
      o[1][tt] = wmma16x16x32(pa1, bv, o[1][tt]);
    }
  }

  // normalize and write [B,S,HID] f16 (heads concatenated)
#pragma unroll
  for (int mi = 0; mi < 2; ++mi)
#pragma unroll
    for (int r = 0; r < 8; ++r) {
      float inv = lrun[mi][r] > 0.f ? 1.f / lrun[mi][r] : 0.f;
      int srow = q0 + mi * 16 + r + 8 * hi;
      size_t base = ((size_t)(b * SEQ + srow)) * HIDDEN + h * HEADD + (lane & 15);
#pragma unroll
      for (int tt = 0; tt < 4; ++tt)
        attn_out[base + tt * 16] = (_Float16)(o[mi][tt][r] * inv);
    }
}

// ---------------------------------------------------------------------------
// classifier on token 0: out[b,c] = h[b,0,:] . Wc[c,:] + bc[c]
// ---------------------------------------------------------------------------
__global__ __launch_bounds__(256)
void cls_kernel(const float* __restrict__ h32, const float* __restrict__ Wc,
                const float* __restrict__ bc, float* __restrict__ out) {
  __shared__ float red[4][256];
  int t = threadIdx.x;
  float acc[4] = {0.f, 0.f, 0.f, 0.f};
  for (int k = t; k < HIDDEN; k += 256) {
    float h0 = h32[k];
    float h1 = h32[(size_t)SEQ * HIDDEN + k];
    float w0 = Wc[k], w1 = Wc[HIDDEN + k];
    acc[0] += h0 * w0; acc[1] += h0 * w1;
    acc[2] += h1 * w0; acc[3] += h1 * w1;
  }
#pragma unroll
  for (int i = 0; i < 4; ++i) red[i][t] = acc[i];
  __syncthreads();
  for (int o = 128; o > 0; o >>= 1) {
    if (t < o)
#pragma unroll
      for (int i = 0; i < 4; ++i) red[i][t] += red[i][t + o];
    __syncthreads();
  }
  if (t < 4) out[t] = red[t][0] + bc[t & 1];
}

// ---------------------------------------------------------------------------
extern "C" void kernel_launch(void* const* d_in, const int* in_sizes, int n_in,
                              void* d_out, int out_size, void* d_ws, size_t ws_size,
                              hipStream_t stream) {
  (void)in_sizes; (void)n_in; (void)out_size; (void)ws_size;
  const int*   input_ids = (const int*)d_in[0];
  const int*   amask     = (const int*)d_in[1];
  const float* wemb = (const float*)d_in[2];
  const float* pemb = (const float*)d_in[3];
  const float* temb = (const float*)d_in[4];
  const float* elw  = (const float*)d_in[5];
  const float* elb  = (const float*)d_in[6];
  const float* Wq = (const float*)d_in[7];   const float* bq = (const float*)d_in[8];
  const float* Wk = (const float*)d_in[9];   const float* bk = (const float*)d_in[10];
  const float* Wv = (const float*)d_in[11];  const float* bv = (const float*)d_in[12];
  const float* Wo = (const float*)d_in[13];  const float* bo = (const float*)d_in[14];
  const float* l1w = (const float*)d_in[15]; const float* l1b = (const float*)d_in[16];
  const float* Wi = (const float*)d_in[17];  const float* bi = (const float*)d_in[18];
  const float* Wf = (const float*)d_in[19];  const float* bf = (const float*)d_in[20];
  const float* l2w = (const float*)d_in[21]; const float* l2b = (const float*)d_in[22];
  const float* Wc = (const float*)d_in[23];  const float* bc = (const float*)d_in[24];

  char* ws = (char*)d_ws;
  size_t off = 0;
  auto carve = [&](size_t bytes) -> void* {
    void* p = ws + off;
    off = (off + bytes + 255) & ~(size_t)255;
    return p;
  };
  int*      pos_ids = (int*)carve((size_t)NTOK * 4);
  float*    h32   = (float*)carve((size_t)NTOK * HIDDEN * 4);
  _Float16* h16   = (_Float16*)carve((size_t)NTOK * HIDDEN * 2);
  _Float16* q16   = (_Float16*)carve((size_t)NTOK * HIDDEN * 2);
  _Float16* k16   = (_Float16*)carve((size_t)NTOK * HIDDEN * 2);
  _Float16* v16   = (_Float16*)carve((size_t)NTOK * HIDDEN * 2);
  _Float16* att16 = (_Float16*)carve((size_t)NTOK * HIDDEN * 2);
  float*    p32   = (float*)carve((size_t)NTOK * HIDDEN * 4);
  _Float16* ff16  = (_Float16*)carve((size_t)NTOK * FFDIM * 2);

  const size_t NPROJ = (size_t)LAYERS * HIDDEN * HIDDEN;   // 7,077,888
  const size_t NFF   = (size_t)LAYERS * FFDIM * HIDDEN;    // 28,311,552
  _Float16* qw16 = (_Float16*)carve(NPROJ * 2);
  _Float16* kw16 = (_Float16*)carve(NPROJ * 2);
  _Float16* vw16 = (_Float16*)carve(NPROJ * 2);
  _Float16* ow16 = (_Float16*)carve(NPROJ * 2);
  _Float16* iw16 = (_Float16*)carve(NFF * 2);
  _Float16* fw16 = (_Float16*)carve(NFF * 2);

  // one-time (per launch) fp32 -> f16 weight conversion; weights stay L2-hot
  cvt_kernel<<<(unsigned)(NPROJ / 2048), 256, 0, stream>>>(Wq, qw16);
  cvt_kernel<<<(unsigned)(NPROJ / 2048), 256, 0, stream>>>(Wk, kw16);
  cvt_kernel<<<(unsigned)(NPROJ / 2048), 256, 0, stream>>>(Wv, vw16);
  cvt_kernel<<<(unsigned)(NPROJ / 2048), 256, 0, stream>>>(Wo, ow16);
  cvt_kernel<<<(unsigned)(NFF   / 2048), 256, 0, stream>>>(Wi, iw16);
  cvt_kernel<<<(unsigned)(NFF   / 2048), 256, 0, stream>>>(Wf, fw16);

  pos_kernel<<<1, 32, 0, stream>>>(amask, pos_ids);
  embed_kernel<<<NTOK, 256, 0, stream>>>(input_ids, pos_ids, wemb, pemb, temb,
                                         elw, elb, h32, h16);

  dim3 gP(HIDDEN / BN, NTOK / BM);   // 768-wide GEMMs
  dim3 gF(FFDIM / BN, NTOK / BM);    // 3072-wide GEMM
  dim3 gA(SEQ / 128, NHEADS, BATCH);

  for (int l = 0; l < LAYERS; ++l) {
    const _Float16* wq = qw16 + (size_t)l * HIDDEN * HIDDEN;
    const _Float16* wk = kw16 + (size_t)l * HIDDEN * HIDDEN;
    const _Float16* wv = vw16 + (size_t)l * HIDDEN * HIDDEN;
    const _Float16* wo = ow16 + (size_t)l * HIDDEN * HIDDEN;
    const _Float16* wi = iw16 + (size_t)l * FFDIM * HIDDEN;
    const _Float16* wf = fw16 + (size_t)l * HIDDEN * FFDIM;

    gemm_kernel<<<gP, 256, 0, stream>>>(h16, wq, bq + l * HIDDEN, NTOK, HIDDEN,
                                        HIDDEN, 3, 0.125f, nullptr, q16);
    gemm_kernel<<<gP, 256, 0, stream>>>(h16, wk, bk + l * HIDDEN, NTOK, HIDDEN,
                                        HIDDEN, 3, 1.0f, nullptr, k16);
    gemm_kernel<<<gP, 256, 0, stream>>>(h16, wv, bv + l * HIDDEN, NTOK, HIDDEN,
                                        HIDDEN, 3, 1.0f, nullptr, v16);
    attn_kernel<<<gA, 128, 0, stream>>>(q16, k16, v16, amask, att16);
    gemm_kernel<<<gP, 256, 0, stream>>>(att16, wo, bo + l * HIDDEN, NTOK, HIDDEN,
                                        HIDDEN, 0, 1.0f, p32, nullptr);
    ln_add_kernel<<<NTOK, 256, 0, stream>>>(h32, p32, l1w + l * HIDDEN,
                                            l1b + l * HIDDEN, h32, h16);
    gemm_kernel<<<gF, 256, 0, stream>>>(h16, wi, bi + l * FFDIM, NTOK, FFDIM,
                                        HIDDEN, 2, 1.0f, nullptr, ff16);
    gemm_kernel<<<gP, 256, 0, stream>>>(ff16, wf, bf + l * HIDDEN, NTOK, HIDDEN,
                                        FFDIM, 0, 1.0f, p32, nullptr);
    ln_add_kernel<<<NTOK, 256, 0, stream>>>(h32, p32, l2w + l * HIDDEN,
                                            l2b + l * HIDDEN, h32, h16);
  }

  cls_kernel<<<1, 256, 0, stream>>>(h32, Wc, bc, (float*)d_out);
}
